// MyModel_86225763434608
// MI455X (gfx1250) — matrix operand order repaired and checked by reading, not compiled, not run on hardware
//
#include <hip/hip_runtime.h>
#include <hip/hip_bf16.h>
#include <math.h>

// Model constants (match reference)
#define B_   2
#define T_   2048
#define D_   1024
#define H_   16
#define F_   4096
#define V_   32000
#define HD_  64
#define M_   (B_ * T_)      // 4096 rows
#define EPS_ 1e-6f

typedef __attribute__((ext_vector_type(16))) __bf16 v16bf;
typedef __attribute__((ext_vector_type(8)))  float  v8f;
typedef unsigned int   u32;
typedef unsigned short u16;
typedef __attribute__((ext_vector_type(4))) u32 u32x4;
typedef __attribute__((ext_vector_type(4))) int v4i;

union Frag {            // 16 bf16 = one WMMA A/B operand (8 VGPRs)
  u32x4 q[2];
  v16bf v;
};

__device__ __forceinline__ u16 f2bf(float f) {
  u32 u = __float_as_uint(f);
  u += 0x7FFFu + ((u >> 16) & 1u);   // round-to-nearest-even
  return (u16)(u >> 16);
}

// CDNA5 async global->LDS copy path (ASYNCcnt-tracked), with sync fallback.
// Probe (round 2 diagnostic): builtin takes (v4i addrspace(1)*, v4i addrspace(3)*,
// imm offset, imm cpol).
#if __has_builtin(__builtin_amdgcn_global_load_async_to_lds_b128) && \
    __has_builtin(__builtin_amdgcn_s_wait_asynccnt)
#define USE_ASYNC_LDS 1
typedef __attribute__((address_space(1))) v4i as1_v4i;
typedef __attribute__((address_space(3))) v4i as3_v4i;
#else
#define USE_ASYNC_LDS 0
#endif

__device__ __forceinline__ void wait_async_keep4() {
#if USE_ASYNC_LDS
  __builtin_amdgcn_s_wait_asynccnt(4);   // keep next tile's 4 copies in flight
#endif
}
__device__ __forceinline__ void wait_async_zero() {
#if USE_ASYNC_LDS
  __builtin_amdgcn_s_wait_asynccnt(0);
#endif
}

// ---------------------------------------------------------------------------
// Embedding + sinusoidal positional encoding -> bf16 activations
// ---------------------------------------------------------------------------
__global__ __launch_bounds__(256) void k_embed(const int* __restrict__ x,
                                               const float* __restrict__ emb,
                                               u16* __restrict__ h) {
  const int row = blockIdx.x;          // 0..M_-1
  const int tok = x[row];
  const int t   = row % T_;
  const float* e = emb + (long)tok * D_;
  for (int d = threadIdx.x; d < D_; d += 256) {
    int   i    = d >> 1;
    float freq = __expf(-(logf(10000.0f)) * (float)(2 * i) / (float)D_);
    float ang  = (float)t * freq;
    float pe   = (d & 1) ? __cosf(ang) : __sinf(ang);
    h[(long)row * D_ + d] = f2bf(e[d] + pe);
  }
}

// ---------------------------------------------------------------------------
// W[K,N] fp32 -> Wt[N,K] bf16  (so GEMM B-frag loads are K-contiguous)
// ---------------------------------------------------------------------------
__global__ __launch_bounds__(256) void k_transpose_bf16(const float* __restrict__ W,
                                                        u16* __restrict__ Wt,
                                                        int K, int N) {
  __shared__ float tile[32][33];
  const int n0 = blockIdx.x * 32, k0 = blockIdx.y * 32;
  const int tx = threadIdx.x & 31, ty = threadIdx.x >> 5;   // 32 x 8
  for (int r = ty; r < 32; r += 8)
    tile[r][tx] = W[(long)(k0 + r) * N + n0 + tx];
  __syncthreads();
  for (int r = ty; r < 32; r += 8)
    Wt[(long)(n0 + r) * K + k0 + tx] = f2bf(tile[tx][r]);
}

// ---------------------------------------------------------------------------
// WMMA bf16 GEMM:  out[M,N] = A[M,K](bf16) @ Wt[N,K]^T(bf16) + bias
//   mode 0: fp32 row-major   mode 1: bf16 row-major
//   mode 2: bf16, V-transposed layout [B,H,HD,T] (for attention P@V operand)
// Tiles: BM=BN=128, BK=32; 8 waves, each computes 32x64 (2x4 WMMA C tiles).
// Double-buffered LDS fed by async global->LDS copies (ASYNCcnt pipeline).
// ---------------------------------------------------------------------------
__global__ __launch_bounds__(256) void k_gemm(const u16* __restrict__ A,
                                              const u16* __restrict__ Wt,
                                              const float* __restrict__ bias,
                                              float* __restrict__ outF,
                                              u16* __restrict__ outB,
                                              int Mdim, int Ndim, int Kdim,
                                              int mode, int relu) {
  __shared__ __align__(16) u16 lA[2][128][40];   // +8 pad keeps 16B alignment
  __shared__ __align__(16) u16 lB[2][128][40];

  const int tid  = threadIdx.x;
  const int lane = tid & 31;
  const int wave = tid >> 5;
  const int wm   = wave >> 1;                 // 0..3
  const int wn   = wave & 1;                  // 0..1
  const int n0   = blockIdx.x * 128;
  const int m0   = blockIdx.y * 128;
  const int lrow = lane & 15;
  const int lhi  = lane >> 4;                 // 0 | 1
  const int kb_a = lhi * 8;                   // A-frag K sub-offset
  const int kb_b = lhi * 16;                  // unused; B-frag loads use kb_a

  // this thread's two 16B staging chunks within the 128x32 tile
  const int chunk0 = tid * 2;
  const int r0  = chunk0 >> 2;
  const int kc0 = (chunk0 & 3) << 3;

  v8f acc[2][4];
  const v8f vzero = {0.f,0.f,0.f,0.f,0.f,0.f,0.f,0.f};
#pragma unroll
  for (int i = 0; i < 2; ++i)
#pragma unroll
    for (int j = 0; j < 4; ++j) acc[i][j] = vzero;

  auto stage = [&](int buf, int kk) {
#pragma unroll
    for (int j = 0; j < 2; ++j) {
      int chunk = tid * 2 + j;
      int r  = chunk >> 2;
      int kc = (chunk & 3) << 3;
      const u16* ga = &A [(long)(m0 + r) * Kdim + kk + kc];
      const u16* gb = &Wt[(long)(n0 + r) * Kdim + kk + kc];
#if USE_ASYNC_LDS
      __builtin_amdgcn_global_load_async_to_lds_b128(
          (as1_v4i*)ga, (as3_v4i*)&lA[buf][r][kc], 0, 0);
      __builtin_amdgcn_global_load_async_to_lds_b128(
          (as1_v4i*)gb, (as3_v4i*)&lB[buf][r][kc], 0, 0);
#else
      *(u32x4*)&lA[buf][r][kc] = *(const u32x4*)ga;
      *(u32x4*)&lB[buf][r][kc] = *(const u32x4*)gb;
#endif
    }
  };

  const int nk = Kdim >> 5;
  stage(0, 0);
  for (int ki = 0; ki < nk; ++ki) {
    const int buf = ki & 1;
    if (ki + 1 < nk) {
      stage(buf ^ 1, (ki + 1) << 5);          // async copy next tile
      if (ki + 2 < nk) {                      // L2 prefetch two tiles ahead
        __builtin_prefetch(&A [(long)(m0 + r0) * Kdim + ((ki + 2) << 5) + kc0], 0, 1);
        __builtin_prefetch(&Wt[(long)(n0 + r0) * Kdim + ((ki + 2) << 5) + kc0], 0, 1);
      }
      wait_async_keep4();                     // current tile's copies done
    } else {
      wait_async_zero();
    }
    __syncthreads();                          // tile visible to all waves

    Frag a[2], b[4];
#pragma unroll
    for (int mt = 0; mt < 2; ++mt) {          // A 16x32 frag per ISA lane map
      int row = wm * 32 + mt * 16 + lrow;
      a[mt].q[0] = *(const u32x4*)&lA[buf][row][kb_a];
      a[mt].q[1] = *(const u32x4*)&lA[buf][row][kb_a + 16];
    }
#pragma unroll
    for (int nt = 0; nt < 4; ++nt) {          // B 32x16 frag: col=lane, K split 8+8 like A
      int col = wn * 64 + nt * 16 + lrow;
      b[nt].q[0] = *(const u32x4*)&lB[buf][col][kb_a];
      b[nt].q[1] = *(const u32x4*)&lB[buf][col][kb_a + 16];
    }
#pragma unroll
    for (int mt = 0; mt < 2; ++mt)
#pragma unroll
      for (int nt = 0; nt < 4; ++nt)
        acc[mt][nt] = __builtin_amdgcn_wmma_f32_16x16x32_bf16(
            false, a[mt].v, false, b[nt].v, (short)0, acc[mt][nt], false, false);
    __syncthreads();                          // done reading before re-stage
  }

  // epilogue: C layout lane=col (within 16), VGPR r -> row r + 8*lhi
#pragma unroll
  for (int mt = 0; mt < 2; ++mt) {
#pragma unroll
    for (int nt = 0; nt < 4; ++nt) {
      int col  = n0 + wn * 64 + nt * 16 + lrow;
      float bb = bias ? bias[col] : 0.0f;
#pragma unroll
      for (int r = 0; r < 8; ++r) {
        int row  = m0 + wm * 32 + mt * 16 + r + 8 * lhi;
        float vv = acc[mt][nt][r] + bb;
        if (relu) vv = fmaxf(vv, 0.0f);
        if (mode == 0) {
          outF[(long)row * Ndim + col] = vv;
        } else if (mode == 1) {
          outB[(long)row * Ndim + col] = f2bf(vv);
        } else {  // V^T layout: [B,H,HD,T]
          int hh = col / HD_, d = col % HD_;
          long idx = (((long)(row / T_) * H_ + hh) * HD_ + d) * T_ + (row % T_);
          outB[idx] = f2bf(vv);
        }
      }
    }
  }
}

// ---------------------------------------------------------------------------
// Flash-style causal attention, one wave per 16-query tile per (b,h).
// scores = Q@K^T / 8 (+1.0 on allowed entries, per reference mask), online
// softmax, O += P@V via WMMA.  q,k: [B*T, D] bf16.  vT: [B,H,HD,T] bf16.
// ---------------------------------------------------------------------------
__global__ __launch_bounds__(128) void k_attn(const u16* __restrict__ q,
                                              const u16* __restrict__ k,
                                              const u16* __restrict__ vT,
                                              u16* __restrict__ o) {
  __shared__ __align__(16) u16 lP[4][16][40];   // per-wave P (16x32) scratch
  const int lane = threadIdx.x & 31;
  const int wave = threadIdx.x >> 5;
  const int gw   = blockIdx.x * 4 + wave;
  const int qt   = gw & (T_ / 16 - 1);          // 0..127
  const int bh   = gw >> 7;
  const int b    = bh >> 4;
  const int h    = bh & (H_ - 1);
  const int lrow = lane & 15;
  const int lhi  = lane >> 4;

  Frag aq[2];                                    // Q 16x64 as two 16x32 A-frags
  {
    int trow = qt * 16 + lrow;
    const u16* qp = q + (long)(b * T_ + trow) * D_ + h * HD_;
#pragma unroll
    for (int s = 0; s < 2; ++s) {
      aq[s].q[0] = *(const u32x4*)&qp[s * 32 + lhi * 8];
      aq[s].q[1] = *(const u32x4*)&qp[s * 32 + lhi * 8 + 16];
    }
  }

  v8f vacc[4];
  const v8f vzero = {0.f,0.f,0.f,0.f,0.f,0.f,0.f,0.f};
#pragma unroll
  for (int d = 0; d < 4; ++d) vacc[d] = vzero;
  float mrun[8], lrun[8];
#pragma unroll
  for (int r = 0; r < 8; ++r) { mrun[r] = -3.0e38f; lrun[r] = 0.0f; }

  const int jmax = (qt * 16 + 15) >> 5;          // key tiles of 32
  for (int j = 0; j <= jmax; ++j) {
    Frag bk[2][2];                               // [k-step][n-tile]
#pragma unroll
    for (int nt = 0; nt < 2; ++nt) {
      int key = j * 32 + nt * 16 + lrow;
      const u16* kp = k + (long)(b * T_ + key) * D_ + h * HD_;
#pragma unroll
      for (int s = 0; s < 2; ++s) {
        bk[s][nt].q[0] = *(const u32x4*)&kp[s * 32 + lhi * 8];
        bk[s][nt].q[1] = *(const u32x4*)&kp[s * 32 + lhi * 8 + 16];
      }
    }
    v8f sc0 = vzero, sc1 = vzero;
#pragma unroll
    for (int s = 0; s < 2; ++s) {
      sc0 = __builtin_amdgcn_wmma_f32_16x16x32_bf16(false, aq[s].v, false, bk[s][0].v,
                                                    (short)0, sc0, false, false);
      sc1 = __builtin_amdgcn_wmma_f32_16x16x32_bf16(false, aq[s].v, false, bk[s][1].v,
                                                    (short)0, sc1, false, false);
    }
    float s0[8], s1[8];
#pragma unroll
    for (int r = 0; r < 8; ++r) {
      int qr   = qt * 16 + r + 8 * lhi;
      int key0 = j * 32 + lrow;
      int key1 = key0 + 16;
      s0[r] = (key0 > qr) ? -3.0e38f : sc0[r] * 0.125f + 1.0f;  // mask adds +1.0
      s1[r] = (key1 > qr) ? -3.0e38f : sc1[r] * 0.125f + 1.0f;
    }
#pragma unroll
    for (int r = 0; r < 8; ++r) {
      float tm = fmaxf(s0[r], s1[r]);
#pragma unroll
      for (int off = 1; off < 16; off <<= 1)     // stays within 16-lane half
        tm = fmaxf(tm, __shfl_xor(tm, off, 32));
      float mnew = fmaxf(mrun[r], tm);
      float corr = __expf(mrun[r] - mnew);
      float p0 = __expf(s0[r] - mnew);
      float p1 = __expf(s1[r] - mnew);
      float rs = p0 + p1;
#pragma unroll
      for (int off = 1; off < 16; off <<= 1)
        rs += __shfl_xor(rs, off, 32);
      lrun[r] = lrun[r] * corr + rs;
      mrun[r] = mnew;
#pragma unroll
      for (int d = 0; d < 4; ++d) vacc[d][r] *= corr;
      lP[wave][r + 8 * lhi][lrow]      = f2bf(p0);   // transpose P via LDS
      lP[wave][r + 8 * lhi][16 + lrow] = f2bf(p1);
    }
    asm volatile("s_wait_dscnt 0" ::: "memory");     // order LDS writes->reads
    Frag ap;
    ap.q[0] = *(const u32x4*)&lP[wave][lrow][lhi * 8];
    ap.q[1] = *(const u32x4*)&lP[wave][lrow][lhi * 8 + 16];
#pragma unroll
    for (int d = 0; d < 4; ++d) {                    // O += P @ V
      int dcol = d * 16 + lrow;
      const u16* vp = vT + ((long)(b * H_ + h) * HD_ + dcol) * T_ + j * 32 + lhi * 8;
      Frag bv;
      bv.q[0] = *(const u32x4*)&vp[0];
      bv.q[1] = *(const u32x4*)&vp[16];
      vacc[d] = __builtin_amdgcn_wmma_f32_16x16x32_bf16(false, ap.v, false, bv.v,
                                                        (short)0, vacc[d], false, false);
    }
  }

#pragma unroll
  for (int r = 0; r < 8; ++r) {
    float inv = 1.0f / lrun[r];
    int trow  = qt * 16 + r + 8 * lhi;
    u16* op = o + (long)(b * T_ + trow) * D_ + h * HD_;
#pragma unroll
    for (int d = 0; d < 4; ++d)
      op[d * 16 + lrow] = f2bf(vacc[d][r] * inv);
  }
}

// ---------------------------------------------------------------------------
// LayerNorm (unbiased std, per reference): out bf16 for next GEMM
// ---------------------------------------------------------------------------
__global__ __launch_bounds__(256) void k_layernorm(const float* __restrict__ x,
                                                   const float* __restrict__ g,
                                                   const float* __restrict__ be,
                                                   u16* __restrict__ out) {
  __shared__ float redA[256], redB[256];
  const int row = blockIdx.x;
  const int tid = threadIdx.x;
  const float* xr = x + (long)row * D_;
  float s = 0.0f, s2 = 0.0f;
  for (int d = tid; d < D_; d += 256) { float v = xr[d]; s += v; s2 += v * v; }
  redA[tid] = s; redB[tid] = s2;
  __syncthreads();
  for (int off = 128; off > 0; off >>= 1) {
    if (tid < off) { redA[tid] += redA[tid + off]; redB[tid] += redB[tid + off]; }
    __syncthreads();
  }
  float mean = redA[0] / (float)D_;
  float var  = (redB[0] - (float)D_ * mean * mean) / (float)(D_ - 1);
  float inv  = 1.0f / (sqrtf(fmaxf(var, 0.0f)) + EPS_);
  for (int d = tid; d < D_; d += 256)
    out[(long)row * D_ + d] = f2bf((xr[d] - mean) * inv * g[d] + be[d]);
}

// ---------------------------------------------------------------------------
// In-place row softmax over V=32000 (operates on d_out)
// ---------------------------------------------------------------------------
__global__ __launch_bounds__(512) void k_softmax(float* __restrict__ logits) {
  __shared__ float red[512];
  const int row = blockIdx.x;
  const int tid = threadIdx.x;
  float* xr = logits + (long)row * V_;
  float mx = -3.0e38f;
  for (int c = tid; c < V_; c += 512) mx = fmaxf(mx, xr[c]);
  red[tid] = mx; __syncthreads();
  for (int off = 256; off > 0; off >>= 1) {
    if (tid < off) red[tid] = fmaxf(red[tid], red[tid + off]);
    __syncthreads();
  }
  mx = red[0];
  __syncthreads();
  float s = 0.0f;
  for (int c = tid; c < V_; c += 512) { float e = __expf(xr[c] - mx); xr[c] = e; s += e; }
  red[tid] = s; __syncthreads();
  for (int off = 256; off > 0; off >>= 1) {
    if (tid < off) red[tid] += red[tid + off];
    __syncthreads();
  }
  float inv = 1.0f / red[0];
  for (int c = tid; c < V_; c += 512) xr[c] *= inv;
}

// ---------------------------------------------------------------------------
// Host-side orchestration
// ---------------------------------------------------------------------------
extern "C" void kernel_launch(void* const* d_in, const int* in_sizes, int n_in,
                              void* d_out, int out_size, void* d_ws, size_t ws_size,
                              hipStream_t stream) {
  (void)in_sizes; (void)n_in; (void)out_size; (void)ws_size;
  const int*   x   = (const int*)  d_in[0];
  const float* emb = (const float*)d_in[1];
  const float* Wq  = (const float*)d_in[2];  const float* bq  = (const float*)d_in[3];
  const float* Wk  = (const float*)d_in[4];  const float* bk  = (const float*)d_in[5];
  const float* Wv  = (const float*)d_in[6];  const float* bv  = (const float*)d_in[7];
  const float* Wo  = (const float*)d_in[8];  const float* bo  = (const float*)d_in[9];
  const float* g1  = (const float*)d_in[10]; const float* be1 = (const float*)d_in[11];
  const float* W1  = (const float*)d_in[12]; const float* bf1 = (const float*)d_in[13];
  const float* W2  = (const float*)d_in[14]; const float* bf2 = (const float*)d_in[15];
  const float* g2  = (const float*)d_in[16]; const float* be2 = (const float*)d_in[17];
  const float* Wl  = (const float*)d_in[18]; const float* bl  = (const float*)d_in[19];
  float* out = (float*)d_out;

  // workspace carving (~207 MB)
  char* ws = (char*)d_ws;
  auto alloc = [&](size_t bytes) -> void* {
    void* p = ws; ws += (bytes + 255) & ~(size_t)255; return p;
  };
  u16* wqT   = (u16*)alloc((size_t)D_ * D_ * 2);
  u16* wkT   = (u16*)alloc((size_t)D_ * D_ * 2);
  u16* wvT   = (u16*)alloc((size_t)D_ * D_ * 2);
  u16* woT   = (u16*)alloc((size_t)D_ * D_ * 2);
  u16* w1T   = (u16*)alloc((size_t)F_ * D_ * 2);
  u16* w2T   = (u16*)alloc((size_t)D_ * F_ * 2);
  u16* wlT   = (u16*)alloc((size_t)V_ * D_ * 2);
  u16* hB    = (u16*)alloc((size_t)M_ * D_ * 2);
  u16* qB    = (u16*)alloc((size_t)M_ * D_ * 2);
  u16* kB    = (u16*)alloc((size_t)M_ * D_ * 2);
  u16* vTB   = (u16*)alloc((size_t)M_ * D_ * 2);
  u16* aoB   = (u16*)alloc((size_t)M_ * D_ * 2);
  float* oF  = (float*)alloc((size_t)M_ * D_ * 4);
  u16* h1B   = (u16*)alloc((size_t)M_ * D_ * 2);
  u16* ffB   = (u16*)alloc((size_t)M_ * F_ * 2);
  float* f2F = (float*)alloc((size_t)M_ * D_ * 4);
  u16* h2B   = (u16*)alloc((size_t)M_ * D_ * 2);

  // weight transposes (fp32 -> bf16 [N,K])
  k_transpose_bf16<<<dim3(D_/32, D_/32), 256, 0, stream>>>(Wq, wqT, D_, D_);
  k_transpose_bf16<<<dim3(D_/32, D_/32), 256, 0, stream>>>(Wk, wkT, D_, D_);
  k_transpose_bf16<<<dim3(D_/32, D_/32), 256, 0, stream>>>(Wv, wvT, D_, D_);
  k_transpose_bf16<<<dim3(D_/32, D_/32), 256, 0, stream>>>(Wo, woT, D_, D_);
  k_transpose_bf16<<<dim3(F_/32, D_/32), 256, 0, stream>>>(W1, w1T, D_, F_);
  k_transpose_bf16<<<dim3(D_/32, F_/32), 256, 0, stream>>>(W2, w2T, F_, D_);
  k_transpose_bf16<<<dim3(V_/32, D_/32), 256, 0, stream>>>(Wl, wlT, D_, V_);

  // embedding + positional encoding
  k_embed<<<M_, 256, 0, stream>>>(x, emb, hB);

  // QKV projections (V stored transposed [B,H,HD,T] for attention)
  dim3 gD(D_/128, M_/128);
  k_gemm<<<gD, 256, 0, stream>>>(hB, wqT, bq, nullptr, qB,  M_, D_, D_, 1, 0);
  k_gemm<<<gD, 256, 0, stream>>>(hB, wkT, bk, nullptr, kB,  M_, D_, D_, 1, 0);
  k_gemm<<<gD, 256, 0, stream>>>(hB, wvT, bv, nullptr, vTB, M_, D_, D_, 2, 0);

  // causal flash attention
  k_attn<<<B_ * H_ * (T_/16) / 4, 128, 0, stream>>>(qB, kB, vTB, aoB);

  // output projection -> fp32 for LN1
  k_gemm<<<gD, 256, 0, stream>>>(aoB, woT, bo, oF, nullptr, M_, D_, D_, 0, 0);
  k_layernorm<<<M_, 256, 0, stream>>>(oF, g1, be1, h1B);

  // FFN
  k_gemm<<<dim3(F_/128, M_/128), 256, 0, stream>>>(h1B, w1T, bf1, nullptr, ffB,
                                                   M_, F_, D_, 1, 1);
  k_gemm<<<gD, 256, 0, stream>>>(ffB, w2T, bf2, f2F, nullptr, M_, D_, F_, 0, 0);
  k_layernorm<<<M_, 256, 0, stream>>>(f2F, g2, be2, h2B);

  // logits straight into d_out, then in-place softmax over V
  k_gemm<<<dim3(V_/128, M_/128), 256, 0, stream>>>(h2B, wlT, bl, out, nullptr,
                                                   M_, V_, D_, 0, 0);
  k_softmax<<<M_, 512, 0, stream>>>(out);
}
